// EmbeddingWithWord_59614146068535
// MI455X (gfx1250) — matrix operand order repaired and checked by reading, not compiled, not run on hardware
//
#include <hip/hip_runtime.h>
#include <hip/hip_bf16.h>
#include <math.h>

typedef __attribute__((ext_vector_type(16))) _Float16 v16h;
typedef __attribute__((ext_vector_type(8)))  float    v8f;

#define SENSE_NUM 80000
#define DIM       100
#define NTOK      52
#define WV_ROWS   100000
#define WV_DIM    300
#define H_STRIDE  112     // padded feature stride (7 N-tiles of 16)
#define E_STRIDE  101     // odd stride -> conflict-free distance phase
#define EPSF      1e-5f

// ---- WMMA f16 A-fragment builders (16x32 tile of row-major M x K) ----------
// ISA layout (CDNA5 ISA): lane m (hi=0) holds K = k0+{0..7, 16..23},
// lane m+16 (hi=1) holds K = k0+{8..15, 24..31}.

// Full chunk: k0+32 <= kmax. Unconditional contiguous loads (vectorizable).
__device__ __forceinline__ v16h a_frag_full(const float* __restrict__ row,
                                            int k0, int hi) {
  v16h a;
  const float* p0 = row + k0 + hi * 8;
  const float* p1 = p0 + 16;
#pragma unroll
  for (int e = 0; e < 8; ++e) {
    a[e]     = (_Float16)p0[e];
    a[8 + e] = (_Float16)p1[e];
  }
  return a;
}

// Tail chunk: loads stay unconditional via index clamp; invalid lanes are
// masked to zero in registers (v_cndmask, no control flow).
__device__ __forceinline__ v16h a_frag_tail(const float* __restrict__ row,
                                            int k0, int hi, int kmax) {
  v16h a;
  const int b0 = k0 + hi * 8;
  const int b1 = b0 + 16;
#pragma unroll
  for (int e = 0; e < 8; ++e) {
    const int ka = b0 + e, kb = b1 + e;
    const float xa = row[ka < kmax ? ka : kmax - 1];
    const float xb = row[kb < kmax ? kb : kmax - 1];
    a[e]     = (ka < kmax) ? (_Float16)xa : (_Float16)0.f;
    a[8 + e] = (kb < kmax) ? (_Float16)xb : (_Float16)0.f;
  }
  return a;
}

// ---- WMMA f16 B-fragment builders (32x16 tile of B = W^T) ------------------
// Lane n (hi=0) holds K = k0+{0..7, 16..23} of column n; lane n+16 holds K = k0+{8..15, 24..31}.
// Column of B == contiguous row of W.

__device__ __forceinline__ v16h b_frag_full(const float* __restrict__ wrow,
                                            int k0, int hi) {
  v16h b;
  const float* p = wrow + k0 + hi * 8;
#pragma unroll
  for (int e = 0; e < 8; ++e) {
    b[e]     = (_Float16)p[e];
    b[8 + e] = (_Float16)p[16 + e];
  }
  return b;
}

__device__ __forceinline__ v16h b_frag_tail(const float* __restrict__ wrow,
                                            int k0, int hi, int kmax) {
  v16h b;
  const int base = k0 + hi * 8;
#pragma unroll
  for (int e = 0; e < 16; ++e) {
    const int k = base + (e < 8 ? e : e + 8);
    const float x = wrow[k < kmax ? k : kmax - 1];
    b[e] = (k < kmax) ? (_Float16)x : (_Float16)0.f;
  }
  return b;
}

__global__ __launch_bounds__(128) void embed_poincare_kernel(
    const int*   __restrict__ inputs,    // (B, 52) token ids
    const float* __restrict__ lt_weight, // (80000, 100)
    const float* __restrict__ wv_emb,    // (100000, 300)
    const float* __restrict__ W1,        // (100, 300)
    const float* __restrict__ b1,        // (100,)
    const float* __restrict__ W2,        // (101, 100)
    const float* __restrict__ b2,        // (101,)
    float*       __restrict__ out)       // (B, 51)
{
  __shared__ float lds_h[4][16][H_STRIDE];   // per-wave MLP tile (reused L1->L2)
  __shared__ float lds_e[NTOK][E_STRIDE];    // final 100-d embeddings

  const int tid   = threadIdx.x;
  const int w     = tid >> 5;        // wave id 0..3 -> 16-token tile
  const int l     = tid & 31;        // lane
  const int m     = l & 15;          // row (A) / col (B,D) within tile
  const int hi    = l >> 4;          // lane half selector
  const int batch = blockIdx.x;

  // Token for this lane's A-matrix row (tail tokens 52..63 run with clamped
  // index; their results are never read).
  const int t  = w * 16 + m;
  const int id = (t < NTOK) ? inputs[batch * NTOK + t] : 0;
  int widx = id - SENSE_NUM;
  widx = widx < 0 ? 0 : (widx > WV_ROWS - 1 ? WV_ROWS - 1 : widx);
  const float* xrow = wv_emb + (size_t)widx * WV_DIM;

  // Clamped weight-row bases: out-of-range output columns duplicate a valid
  // column; those D slots are never consumed downstream.
  const int colm = m;  // column this lane owns within each N-tile

  // ---------------- layer 1: H = tanh(X (16x300) @ W1^T + b1) ---------------
  v8f acc[7];
#pragma unroll
  for (int nt = 0; nt < 7; ++nt)
    acc[nt] = (v8f){0.f, 0.f, 0.f, 0.f, 0.f, 0.f, 0.f, 0.f};

  for (int k0 = 0; k0 < 288; k0 += 32) {              // 9 full K-chunks
    v16h a = a_frag_full(xrow, k0, hi);
#pragma unroll
    for (int nt = 0; nt < 7; ++nt) {
      int col = nt * 16 + colm;
      col = col < DIM ? col : DIM - 1;
      v16h bm = b_frag_full(W1 + (size_t)col * WV_DIM, k0, hi);
      acc[nt] = __builtin_amdgcn_wmma_f32_16x16x32_f16(
          false, a, false, bm, (short)0, acc[nt], false, false);
    }
  }
  {                                                   // tail chunk k = 288..299
    v16h a = a_frag_tail(xrow, 288, hi, WV_DIM);
#pragma unroll
    for (int nt = 0; nt < 7; ++nt) {
      int col = nt * 16 + colm;
      col = col < DIM ? col : DIM - 1;
      v16h bm = b_frag_tail(W1 + (size_t)col * WV_DIM, 288, hi, WV_DIM);
      acc[nt] = __builtin_amdgcn_wmma_f32_16x16x32_f16(
          false, a, false, bm, (short)0, acc[nt], false, false);
    }
  }

#pragma unroll
  for (int nt = 0; nt < 7; ++nt) {
    const int f = nt * 16 + m;                        // feature index (D layout)
    const int fc = f < DIM ? f : DIM - 1;
    const float bias = b1[fc];
#pragma unroll
    for (int j = 0; j < 8; ++j)                       // rows j and j+8
      lds_h[w][j + hi * 8][f] = tanhf(acc[nt][j] + bias);
  }
  // Intra-wave LDS RAW: DS ops complete in order per wave; the wait also
  // fences compiler reordering of the D-layout stores vs A-layout reloads.
  asm volatile("s_wait_dscnt 0" ::: "memory");

  // ---------------- layer 2: Y = tanh(H (16x100) @ W2^T + b2) ---------------
  v8f acc2[7];
#pragma unroll
  for (int nt = 0; nt < 7; ++nt)
    acc2[nt] = (v8f){0.f, 0.f, 0.f, 0.f, 0.f, 0.f, 0.f, 0.f};

  const float* hrow = &lds_h[w][m][0];
#pragma unroll
  for (int k0 = 0; k0 < 96; k0 += 32) {               // 3 full K-chunks
    v16h a = a_frag_full(hrow, k0, hi);
#pragma unroll
    for (int nt = 0; nt < 7; ++nt) {
      int col = nt * 16 + colm;
      col = col < DIM + 1 ? col : DIM;
      v16h bm = b_frag_full(W2 + (size_t)col * DIM, k0, hi);
      acc2[nt] = __builtin_amdgcn_wmma_f32_16x16x32_f16(
          false, a, false, bm, (short)0, acc2[nt], false, false);
    }
  }
  {                                                   // tail chunk k = 96..99
    v16h a = a_frag_tail(hrow, 96, hi, DIM);
#pragma unroll
    for (int nt = 0; nt < 7; ++nt) {
      int col = nt * 16 + colm;
      col = col < DIM + 1 ? col : DIM;
      v16h bm = b_frag_tail(W2 + (size_t)col * DIM, 96, hi, DIM);
      acc2[nt] = __builtin_amdgcn_wmma_f32_16x16x32_f16(
          false, a, false, bm, (short)0, acc2[nt], false, false);
    }
  }
  asm volatile("s_wait_dscnt 0" ::: "memory");

#pragma unroll
  for (int nt = 0; nt < 7; ++nt) {
    const int f = nt * 16 + m;
    const int fc = f < DIM + 1 ? f : DIM;
    const float bias = b2[fc];
#pragma unroll
    for (int j = 0; j < 8; ++j)
      lds_h[w][j + hi * 8][f] = tanhf(acc2[nt][j] + bias);  // y2 tile, reuse lds_h
  }
  __syncthreads();

  // ------------- select sense/word embedding, renormalize into lds_e --------
  if (tid < NTOK) {
    const int tk  = tid;
    const int id2 = inputs[batch * NTOK + tk];
    if (id2 < SENSE_NUM) {
      int sidx = id2 < 0 ? 0 : (id2 > SENSE_NUM - 1 ? SENSE_NUM - 1 : id2);
      const float* srow = lt_weight + (size_t)sidx * DIM;
      float ss = 0.f;
      for (int k = 0; k < DIM; ++k) { float x = srow[k]; ss += x * x; }
      const float sn    = sqrtf(ss);
      const float scale = fminf(1.f, 1.f / fmaxf(sn, 1e-7f));   // MAX_NORM = 1
      for (int k = 0; k < DIM; ++k) lds_e[tk][k] = srow[k] * scale;
    } else {
      const float* yrow = &lds_h[tk >> 4][tk & 15][0];
      float ss = 0.f;
      for (int k = 0; k < DIM; ++k) { float x = yrow[k]; ss += x * x; }
      const float radius = yrow[DIM];                 // column 100
      const float inv    = radius / sqrtf(ss);
      for (int k = 0; k < DIM; ++k) lds_e[tk][k] = yrow[k] * inv;
    }
  }
  __syncthreads();

  // ------------- Poincare distance: token 0 vs tokens 1..51 -----------------
  if (tid < NTOK - 1) {
    const float* u = &lds_e[0][0];
    const float* v = &lds_e[tid + 1][0];
    float u2 = 0.f, v2 = 0.f, d2 = 0.f;
    for (int k = 0; k < DIM; ++k) {
      const float uk = u[k], vk = v[k];
      u2 += uk * uk;
      v2 += vk * vk;
      const float d = uk - vk;
      d2 += d * d;
    }
    u2 = fminf(fmaxf(u2, 0.f), 1.f - EPSF);
    v2 = fminf(fmaxf(v2, 0.f), 1.f - EPSF);
    float x = 1.f + 2.f * d2 / ((1.f - u2) * (1.f - v2));
    x = fmaxf(x, 1.f + EPSF);
    out[(size_t)batch * (NTOK - 1) + tid] = logf(x + sqrtf(x * x - 1.f));
  }
}

extern "C" void kernel_launch(void* const* d_in, const int* in_sizes, int n_in,
                              void* d_out, int out_size, void* d_ws, size_t ws_size,
                              hipStream_t stream) {
  const int*   inputs = (const int*)d_in[0];
  const float* lt     = (const float*)d_in[1];
  const float* wv     = (const float*)d_in[2];
  const float* W1     = (const float*)d_in[3];
  const float* b1     = (const float*)d_in[4];
  const float* W2     = (const float*)d_in[5];
  const float* b2     = (const float*)d_in[6];
  float*       out    = (float*)d_out;

  const int nbatch = in_sizes[0] / NTOK;   // 8192
  embed_poincare_kernel<<<nbatch, 128, 0, stream>>>(inputs, lt, wv, W1, b1, W2, b2, out);
  (void)n_in; (void)out_size; (void)d_ws; (void)ws_size;
}
